// GCNModelVAE_74380243632355
// MI455X (gfx1250) — matrix operand order repaired and checked by reading, not compiled, not run on hardware
//
#include <hip/hip_runtime.h>
#include <hip/hip_bf16.h>

// ---------------------------------------------------------------------------
// CDNA5 (gfx1250) GCN-VAE forward.
//   - activations & weights pre-converted to bf16 (weights also transposed so
//     both WMMA fragments are K-contiguous -> global_load_b128 only)
//   - support = act @ W via v_wmma_f32_16x16x32_bf16 (f32 accumulate)
//   - out[d] += w_e * support[s] via global_atomic_add_f32 scatter
// N = 10000 nodes (625*16), E = 160000 edges; all dims multiples of 16/32.
// ---------------------------------------------------------------------------

typedef __attribute__((ext_vector_type(16))) __bf16 v16bf;
typedef __attribute__((ext_vector_type(8)))  __bf16 v8bf;
typedef __attribute__((ext_vector_type(8)))  float  v8f;

static __device__ __forceinline__ unsigned short bfbits(float f) {
    // round-to-nearest-even fp32 -> bf16 bits
    unsigned u = __builtin_bit_cast(unsigned, f);
    u += 0x7FFFu + ((u >> 16) & 1u);
    return (unsigned short)(u >> 16);
}

// ---------------------------------------------------------------------------
// GEMM: C[M,N] = A[M,K] @ B[K,N], A bf16 row-major, Bt = B^T bf16 [N,K].
// One wave -> 16x32 C strip (A fragment reused across 2 WMMAs).
// Block = 128 threads = 4 waves = 128 columns. grid.x = N/128, grid.y = M/16.
// Fragment layouts per CDNA5 ISA 7.12.2 (wave32):
//   A 16x32 bf16 : lane&15 = row; lane>>4 = g; VGPR0-3 hold K = 8g+{0..7},
//                  VGPR4-7 hold K = 16+8g+{0..7}  (two b128 loads).
//   B 32x16 bf16 : lane&15 = column; same K order as A: K = 8g+{0..7},
//                  then K = 16+8g+{0..7} -> two b128 loads
//                  from Bt at k0+8g and k0+16+8g.
//   C/D 16x16 f32: col = lane&15, VGPR r -> row r + 8*(lane>>4).
// ---------------------------------------------------------------------------
__global__ __launch_bounds__(128) void gcn_gemm_bf16(
    const __bf16* __restrict__ A,   // [M,K]
    const __bf16* __restrict__ Bt,  // [N,K]  (B transposed)
    float* __restrict__ C, int M, int N, int K)
{
    const int lane  = threadIdx.x & 31;
    const int wave  = threadIdx.x >> 5;
    const int mTile = blockIdx.y << 4;
    const int nTile = ((blockIdx.x << 2) + wave) << 5;   // 32 cols per wave

    const int r = lane & 15;   // row (A) / column (B,C)
    const int g = lane >> 4;   // K-group selector

    v8f acc0 = {}, acc1 = {};
    const __bf16* __restrict__ Ap  = A  + (size_t)(mTile + r) * K + (g << 3);
    const __bf16* __restrict__ Bp0 = Bt + (size_t)(nTile + r) * K + (g << 3);
    const __bf16* __restrict__ Bp1 = Bp0 + (size_t)16 * K;

    for (int k0 = 0; k0 < K; k0 += 32) {
        // A: K = k0 + 8g + {0..7}  and  k0 + 16 + 8g + {0..7}
        const v8bf alo  = *(const v8bf*)(Ap  + k0);
        const v8bf ahi  = *(const v8bf*)(Ap  + k0 + 16);
        // B: K = k0 + 8g + {0..7}  and  k0 + 16 + 8g + {0..7}
        const v8bf b0lo = *(const v8bf*)(Bp0 + k0);
        const v8bf b0hi = *(const v8bf*)(Bp0 + k0 + 16);
        const v8bf b1lo = *(const v8bf*)(Bp1 + k0);
        const v8bf b1hi = *(const v8bf*)(Bp1 + k0 + 16);

        const v16bf a  = __builtin_shufflevector(alo,  ahi,  0,1,2,3,4,5,6,7,8,9,10,11,12,13,14,15);
        const v16bf b0 = __builtin_shufflevector(b0lo, b0hi, 0,1,2,3,4,5,6,7,8,9,10,11,12,13,14,15);
        const v16bf b1 = __builtin_shufflevector(b1lo, b1hi, 0,1,2,3,4,5,6,7,8,9,10,11,12,13,14,15);

        acc0 = __builtin_amdgcn_wmma_f32_16x16x32_bf16(
            false, a, false, b0, (short)0, acc0, false, false);
        acc1 = __builtin_amdgcn_wmma_f32_16x16x32_bf16(
            false, a, false, b1, (short)0, acc1, false, false);
    }

    float* __restrict__ C0 = C + (size_t)(mTile + (g << 3)) * N + nTile + r;
#pragma unroll
    for (int v = 0; v < 8; ++v) {
        C0[(size_t)v * N]      = acc0[v];
        C0[(size_t)v * N + 16] = acc1[v];
    }
}

// ---------------------------------------------------------------------------
// fp32 -> bf16 elementwise (activations). 4 elements/thread.
// ---------------------------------------------------------------------------
__global__ __launch_bounds__(256) void cvt_f32_bf16(
    const float* __restrict__ in, unsigned short* __restrict__ out, long long n4)
{
    const long long t = (long long)blockIdx.x * blockDim.x + threadIdx.x;
    if (t >= n4) return;
    const float4 v = ((const float4*)in)[t];
    ushort4 o;
    o.x = bfbits(v.x); o.y = bfbits(v.y); o.z = bfbits(v.z); o.w = bfbits(v.w);
    ((ushort4*)out)[t] = o;
}

// ---------------------------------------------------------------------------
// fp32 -> bf16 with transpose (weights): W[K,N] -> Wt[N,K].
// ---------------------------------------------------------------------------
__global__ __launch_bounds__(256) void cvt_tr_f32_bf16(
    const float* __restrict__ W, unsigned short* __restrict__ Wt, int K, int N)
{
    const int t = blockIdx.x * blockDim.x + threadIdx.x;
    if (t >= K * N) return;
    const int n = t / K, k = t - n * K;          // coalesced writes to Wt
    Wt[t] = bfbits(W[(size_t)k * N + n]);
}

// ---------------------------------------------------------------------------
// Edge scatter: out[dst[e], :] += w[e] * sup[src[e], :]
// One thread per (edge, 4-feature chunk); D is a power of two (128/256/512).
// ---------------------------------------------------------------------------
__global__ __launch_bounds__(256) void gcn_scatter(
    const float* __restrict__ sup, const int* __restrict__ src,
    const int* __restrict__ dst, const float* __restrict__ w,
    float* __restrict__ out, int E, int D, int chunksLog)
{
    const long long t = (long long)blockIdx.x * blockDim.x + threadIdx.x;
    if (t >= ((long long)E << chunksLog)) return;
    const int e = (int)(t >> chunksLog);
    const int c = ((int)t & ((1 << chunksLog) - 1)) << 2;

    const int   s  = src[e];
    const int   d  = dst[e];
    const float we = w[e];

    const float4 v = *(const float4*)(sup + (size_t)s * D + c);
    float* __restrict__ o = out + (size_t)d * D + c;
    atomicAdd(o + 0, v.x * we);
    atomicAdd(o + 1, v.y * we);
    atomicAdd(o + 2, v.z * we);
    atomicAdd(o + 3, v.w * we);
}

__global__ __launch_bounds__(256) void gcn_relu(float* __restrict__ p, long long n)
{
    const long long t = (long long)blockIdx.x * blockDim.x + threadIdx.x;
    if (t < n) p[t] = fmaxf(p[t], 0.0f);
}

// ---------------------------------------------------------------------------
// Host-side helpers
// ---------------------------------------------------------------------------
static void launch_gemm(const __bf16* A, const __bf16* Bt, float* C,
                        int M, int N, int K, hipStream_t stream)
{
    dim3 grid(N / 128, M / 16);
    gcn_gemm_bf16<<<grid, 128, 0, stream>>>(A, Bt, C, M, N, K);
}

static void launch_cvt(const float* in, unsigned short* out, long long n,
                       hipStream_t stream)
{
    const long long n4 = n >> 2;
    cvt_f32_bf16<<<(int)((n4 + 255) / 256), 256, 0, stream>>>(in, out, n4);
}

static void launch_cvt_tr(const float* W, unsigned short* Wt, int K, int N,
                          hipStream_t stream)
{
    const int n = K * N;
    cvt_tr_f32_bf16<<<(n + 255) / 256, 256, 0, stream>>>(W, Wt, K, N);
}

static void launch_agg(const float* sup, const int* esrc, const int* edst,
                       const float* ew, float* dstbuf, int E, int D, int Nn,
                       bool relu, hipStream_t stream)
{
    hipMemsetAsync(dstbuf, 0, (size_t)Nn * D * sizeof(float), stream);
    const int chunksLog = (D == 512) ? 7 : (D == 256 ? 6 : 5);
    const long long total = (long long)E << chunksLog;
    gcn_scatter<<<(int)((total + 255) / 256), 256, 0, stream>>>(
        sup, esrc, edst, ew, dstbuf, E, D, chunksLog);
    if (relu) {
        const long long n = (long long)Nn * D;
        gcn_relu<<<(int)((n + 255) / 256), 256, 0, stream>>>(dstbuf, n);
    }
}

extern "C" void kernel_launch(void* const* d_in, const int* in_sizes, int n_in,
                              void* d_out, int out_size, void* d_ws, size_t ws_size,
                              hipStream_t stream)
{
    (void)n_in; (void)out_size; (void)ws_size;

    const float* x   = (const float*)d_in[0];   // [N,512]
    const float* ew  = (const float*)d_in[1];   // [E]
    const float* W1  = (const float*)d_in[2];   // [512,256]
    const float* W2  = (const float*)d_in[3];   // [256,128]
    const float* W3  = (const float*)d_in[4];   // [256,128]
    const float* Wd1 = (const float*)d_in[5];   // [128,256]
    const float* Wd2 = (const float*)d_in[6];   // [256,512]
    const int*  esrc = (const int*)d_in[7];     // [E]
    const int*  edst = (const int*)d_in[8];     // [E]

    const int Nn = 10000;
    const int E  = in_sizes[7];

    float* out    = (float*)d_out;
    float* pred   = out;                           // [N,512]
    float* mu     = out + (size_t)Nn * 512;        // [N,128]
    float* logvar = mu  + (size_t)Nn * 128;        // [N,128]

    // --- workspace carve-up (bytes) ---
    char* ws = (char*)d_ws;
    float*          sup  = (float*)ws;                               // 20,480,000 B
    float*          h    = (float*)(ws + 20480000);                  // 10,240,000 B
    unsigned short* actb = (unsigned short*)(ws + 30720000);         // 10,240,000 B
    unsigned short* W1t  = (unsigned short*)(ws + 40960000);         // 512*256*2
    unsigned short* W2t  = W1t + 512 * 256;
    unsigned short* W3t  = W2t + 256 * 128;
    unsigned short* Wd1t = W3t + 256 * 128;
    unsigned short* Wd2t = Wd1t + 128 * 256;

    // --- weight conversion + transpose (tiny, once per call) ---
    launch_cvt_tr(W1,  W1t,  512, 256, stream);
    launch_cvt_tr(W2,  W2t,  256, 128, stream);
    launch_cvt_tr(W3,  W3t,  256, 128, stream);
    launch_cvt_tr(Wd1, Wd1t, 128, 256, stream);
    launch_cvt_tr(Wd2, Wd2t, 256, 512, stream);

    // h1 = relu(agg(x @ W1))
    launch_cvt(x, actb, (long long)Nn * 512, stream);
    launch_gemm((const __bf16*)actb, (const __bf16*)W1t, sup, Nn, 256, 512, stream);
    launch_agg(sup, esrc, edst, ew, h, E, 256, Nn, /*relu=*/true, stream);

    // mu = agg(h1 @ W2); logvar = agg(h1 @ W3)
    launch_cvt(h, actb, (long long)Nn * 256, stream);
    launch_gemm((const __bf16*)actb, (const __bf16*)W2t, sup, Nn, 128, 256, stream);
    launch_agg(sup, esrc, edst, ew, mu, E, 128, Nn, false, stream);
    launch_gemm((const __bf16*)actb, (const __bf16*)W3t, sup, Nn, 128, 256, stream);
    launch_agg(sup, esrc, edst, ew, logvar, E, 128, Nn, false, stream);

    // d1 = relu(agg(mu @ Wd1))   (z = mu, eval mode)
    launch_cvt(mu, actb, (long long)Nn * 128, stream);
    launch_gemm((const __bf16*)actb, (const __bf16*)Wd1t, sup, Nn, 256, 128, stream);
    launch_agg(sup, esrc, edst, ew, h, E, 256, Nn, /*relu=*/true, stream);

    // pred = agg(d1 @ Wd2)
    launch_cvt(h, actb, (long long)Nn * 256, stream);
    launch_gemm((const __bf16*)actb, (const __bf16*)Wd2t, sup, Nn, 512, 256, stream);
    launch_agg(sup, esrc, edst, ew, pred, E, 512, Nn, false, stream);
}
